// RNNSurrogate_69088843923958
// MI455X (gfx1250) — matrix operand order repaired and checked by reading, not compiled, not run on hardware
//
#include <hip/hip_runtime.h>

// Problem dims (fixed by the reference)
#define B_   512
#define T_   512
#define F_   128
#define U_   256
#define O_   32
#define G_   1024            // 4*U
#define K_   384             // F + U  (packed [xt | h] contraction dim)

typedef __attribute__((ext_vector_type(16))) __bf16 v16bf;
typedef __attribute__((ext_vector_type(8)))  float  v8f;
typedef __attribute__((ext_vector_type(4)))  unsigned int u32x4;   // plain builtin vector
typedef __attribute__((ext_vector_type(4)))  unsigned int v4ui;
typedef __attribute__((ext_vector_type(8)))  int v8i_;
typedef __attribute__((ext_vector_type(4)))  int v4i_;

// Global (addrspace(1)) pointer types: force global_load/global_store lowering
// (generic pointers would lower to FLAT, which ties up DScnt as well as LOADcnt).
typedef __attribute__((address_space(1))) const unsigned short gus;
typedef __attribute__((address_space(1))) const u32x4 gu4;
typedef __attribute__((address_space(1))) float gfl;

union Frag16 { unsigned int u[8]; u32x4 q[2]; v16bf v; };
union Frag32 { float f[8]; v8f v; };

#define HAVE_TDM __has_builtin(__builtin_amdgcn_tensor_load_to_lds)

__device__ __forceinline__ unsigned short f2bf(float x) {
  unsigned int u = __float_as_uint(x);
  u += 0x7FFFu + ((u >> 16) & 1u);   // round-to-nearest-even
  return (unsigned short)(u >> 16);
}
__device__ __forceinline__ float fsig(float x) { return 1.0f / (1.0f + __expf(-x)); }
__device__ __forceinline__ float ftanh_(float x) { return 2.0f * fsig(2.0f * x) - 1.0f; }

// ---- Weight repack: Wc[n][k] = bf16( k<F ? Wx[k][n] : Wh[k-F][n] ), K-major rows
__global__ void pack_wc_kernel(const float* __restrict__ Wx,
                               const float* __restrict__ Wh,
                               unsigned short* __restrict__ Wc) {
  int idx = blockIdx.x * 256 + threadIdx.x;
  if (idx >= G_ * K_) return;
  int nn = idx / K_;
  int kk = idx - nn * K_;
  float v = (kk < F_) ? Wx[kk * G_ + nn] : Wh[(kk - F_) * G_ + nn];
  Wc[idx] = f2bf(v);
}

// ---- Wot[n][k] = bf16(Wo[k][n]), K-major rows
__global__ void pack_wo_kernel(const float* __restrict__ Wo,
                               unsigned short* __restrict__ Wot) {
  int idx = blockIdx.x * 256 + threadIdx.x;
  if (idx >= O_ * U_) return;
  int nn = idx / U_;
  int kk = idx - nn * U_;
  Wot[idx] = f2bf(Wo[kk * O_ + nn]);
}

// ---- One-time x -> bf16 (removes per-step converts from the serial scan chain)
__global__ void pack_x_kernel(const float* __restrict__ x,
                              unsigned short* __restrict__ xb) {
  size_t idx = (size_t)blockIdx.x * 256 + threadIdx.x;       // one thread per 4 floats
  if (idx >= (size_t)B_ * T_ * F_ / 4) return;
  float4 p = ((const float4*)x)[idx];
  uint2 o;
  o.x = (unsigned)f2bf(p.x) | ((unsigned)f2bf(p.y) << 16);
  o.y = (unsigned)f2bf(p.z) | ((unsigned)f2bf(p.w) << 16);
  ((uint2*)xb)[idx] = o;
}

#if HAVE_TDM
// TDM 2D tile load: 16 rows x 128 bf16 from xbf (row stride T_*F_ elems) into the
// LDS A buffer (at LDS offset 0) with row padding 128 bf16->384-col layout:
// pad_interval=5 (pad after 64 dwords = 256B), pad_amount=127 (128 dwords = 512B).
// 6-arg clang-23 form: (uint32x4 g0, int32x8 g1, int32x4, int32x4, int32x8, i32 cpol)
__device__ __forceinline__ void tdm_load_x_tile(const unsigned short* xbf, int b0, int t) {
  unsigned long long ga = (unsigned long long)xbf +
                          (((unsigned long long)b0 * T_ + (unsigned)t) * F_) * 2ull;
  v4ui g0;
  g0[0] = 1u;                                   // count=1, is_restore=0, gather off
  g0[1] = 0u;                                   // lds_addr = 0 (A is first LDS alloc)
  g0[2] = (unsigned)ga;                         // global_addr[31:0]
  g0[3] = ((unsigned)(ga >> 32) & 0x1FFFFFFu) | 0x80000000u;  // addr[56:32] | type=2
  v8i_ g1;
  g1[0] = (int)(0x10000u            /* data_size=2B */
              | 0x100000u           /* pad_enable   */
              | (5u << 22)          /* pad_interval: 64 dwords */
              | (127u << 25));      /* pad_amount: 128 dwords  */
  g1[1] = (int)(128u << 16);        // tensor_dim0 = 128
  g1[2] = (int)(16u << 16);         // tensor_dim0 hi=0 | tensor_dim1 = 16
  g1[3] = (int)(128u << 16);        // tensor_dim1 hi=0 | tile_dim0 = 128
  g1[4] = 16;                        // tile_dim1 = 16, tile_dim2 = 0
  g1[5] = T_ * F_;                   // tensor_dim0_stride[31:0] = 65536 elems
  g1[6] = 0; g1[7] = 0;
  v4i_ z4 = {0, 0, 0, 0};
  v8i_ z8 = {0, 0, 0, 0, 0, 0, 0, 0};
  __builtin_amdgcn_tensor_load_to_lds(g0, g1, z4, z4, z8, 0);
}
#endif

// ---- Persistent per-batch-tile LSTM scan.
// One block = 16 batch rows, 8 waves. Wave w owns z N-tiles {w+8j}, j=0..7,
// so i/f/g/o of a hidden unit land in the same lane -> cell state c stays in VGPRs.
__global__ __launch_bounds__(256, 1)
void lstm_scan_kernel(const unsigned short* __restrict__ xbf,  // [B,T,F] bf16
                      const unsigned short* __restrict__ Wc,   // [G_][K_] bf16
                      const unsigned short* __restrict__ Wot,  // [O_][U_] bf16
                      const float* __restrict__ b,
                      const float* __restrict__ bo,
                      float* __restrict__ out) {
  __shared__ __align__(16) unsigned short A[16 * K_];   // [m][ xt(0..127) | h(128..383) ]

  const int tid  = threadIdx.x;
  const int wv   = tid >> 5;
  const int lane = tid & 31;
  const int n    = lane & 15;
  const int half = lane >> 4;
  const int b0   = blockIdx.x * 16;

  // h(0) = 0
  for (int idx = tid; idx < 16 * U_; idx += 256)
    A[(idx >> 8) * K_ + F_ + (idx & 255)] = (unsigned short)0;

#if HAVE_TDM
  if (wv == 0) tdm_load_x_tile(xbf, b0, 0);   // prologue: stage x(0) via TDM
#endif

  // per-lane gate biases (z-col = (wv+8j)*16 + n), loop-invariant
  float bias[8];
  #pragma unroll
  for (int j = 0; j < 8; ++j) bias[j] = b[(wv + 8 * j) * 16 + n];
  float biasy = 0.0f;
  if (wv < 2) biasy = bo[wv * 16 + n];

  // persistent cell state: c for (m = r+8*half, u = wv*16 + j0*128 + n)
  float cfrag[2][8];
  #pragma unroll
  for (int j0 = 0; j0 < 2; ++j0)
    #pragma unroll
    for (int r = 0; r < 8; ++r) cfrag[j0][r] = 0.0f;

  const int arow = (lane & 15) * K_;              // A-operand LDS row (m = lane&15)
  const int wzoff = (wv * 16 + n) * K_ + half * 8;   // z-weight lane offset (elems)
  const int wyoff = (wv * 16 + n) * U_ + half * 8;   // y-weight lane offset (elems)
  gfl* outg = (gfl*)out;

  for (int t = 0; t < T_; ++t) {
    // Anti-LICM: opaque zero byte-offset, renewed every iteration, so weight
    // fragment loads cannot be hoisted/CSE'd across the time loop (they must
    // STREAM from L2 each step; hoisting 3KB/lane of B-fragments spills).
    // Offset-only opacity keeps the base pointers' addrspace -> global_load.
    size_t woff = 0;
    asm volatile("" : "+s"(woff));
    const gus* wz = (const gus*)(Wc + woff) + wzoff;
    const gus* wy = (const gus*)(Wot + woff) + wyoff;

#if HAVE_TDM
    if (wv == 0) __builtin_amdgcn_s_wait_tensorcnt(0);  // x(t) tile landed in LDS
#else
    { // fallback: cooperative bf16 tile copy (4KB), 16B per thread
      int m = tid >> 4, c = (tid & 15) * 8;
      *(u32x4*)&A[m * K_ + c] =
          *(const u32x4*)(xbf + ((size_t)(b0 + m) * T_ + t) * F_ + c);
    }
#endif
    __syncthreads();   // xt and h(t-1) visible

    // ---- z = [xt|h] @ Wc + b   (M=16, N=1024, K=384; wave does 8 strided N-tiles)
    Frag32 acc[8];
    #pragma unroll
    for (int j = 0; j < 8; ++j)
      #pragma unroll
      for (int r = 0; r < 8; ++r) acc[j].f[r] = bias[j];

    for (int kk = 0; kk < 12; ++kk) {
      const int kb = kk * 32;
      Frag16 af;   // ISA A layout: VGPR v holds k = (v>>2)*16 + half*8 + (v&3)*2
      af.q[0] = *(const u32x4*)&A[arow + kb + half * 8];
      af.q[1] = *(const u32x4*)&A[arow + kb + 16 + half * 8];
      #pragma unroll
      for (int j = 0; j < 8; ++j) {
        // j stride = 8 N-tiles = 128 rows * K_ elems = 98304 B (fits IOFFSET imm)
        const gus* wp = wz + j * (128 * K_) + kb;
        Frag16 bf;   // K-major packed row: k = half*8 + [0..7] and 16 + half*8 + [0..7]
        bf.q[0] = *(const gu4*)wp;
        bf.q[1] = *(const gu4*)(wp + 16);
        acc[j].v = __builtin_amdgcn_wmma_f32_16x16x32_bf16(
            false, af.v, false, bf.v, (short)0, acc[j].v, false, false);
      }
    }
    __syncthreads();   // all waves done reading A before h(t)/x(t+1) overwrite

#if HAVE_TDM
    // x region of A is now dead for this step: prefetch x(t+1) via TDM, fully
    // overlapped with gate math + y projection below.
    if (wv == 0 && t + 1 < T_) tdm_load_x_tile(xbf, b0, t + 1);
#endif

    // ---- gates entirely in registers; c persistent; write h(t) bf16 into A
    #pragma unroll
    for (int j0 = 0; j0 < 2; ++j0) {
      #pragma unroll
      for (int r = 0; r < 8; ++r) {
        float iv = acc[j0].f[r];
        float fv = acc[j0 + 2].f[r];
        float gv = acc[j0 + 4].f[r];
        float ov = acc[j0 + 6].f[r];
        float cc = fsig(fv) * cfrag[j0][r] + fsig(iv) * ftanh_(gv);
        cfrag[j0][r] = cc;
        float hh = fsig(ov) * ftanh_(cc);
        A[(r + 8 * half) * K_ + F_ + wv * 16 + j0 * 128 + n] = f2bf(hh);
      }
    }
    __syncthreads();   // h(t) visible

    // ---- y = h @ Wo + bo  (N=32 -> waves 0,1; K=256)
    if (wv < 2) {
      Frag32 ay;
      #pragma unroll
      for (int r = 0; r < 8; ++r) ay.f[r] = biasy;
      #pragma unroll
      for (int kk = 0; kk < 8; ++kk) {
        const int kb = kk * 32;
        Frag16 af;
        af.q[0] = *(const u32x4*)&A[arow + F_ + kb + half * 8];
        af.q[1] = *(const u32x4*)&A[arow + F_ + kb + 16 + half * 8];
        const gus* wp = wy + kb;
        Frag16 bf;
        bf.q[0] = *(const gu4*)wp;
        bf.q[1] = *(const gu4*)(wp + 16);
        ay.v = __builtin_amdgcn_wmma_f32_16x16x32_bf16(
            false, af.v, false, bf.v, (short)0, ay.v, false, false);
      }
      #pragma unroll
      for (int r = 0; r < 8; ++r)
        outg[((size_t)(b0 + r + 8 * half) * T_ + t) * O_ + wv * 16 + n] = ay.f[r];
    }
  }
}

extern "C" void kernel_launch(void* const* d_in, const int* in_sizes, int n_in,
                              void* d_out, int out_size, void* d_ws, size_t ws_size,
                              hipStream_t stream) {
  const float* x  = (const float*)d_in[0];  // [B,T,F]
  const float* Wx = (const float*)d_in[1];  // [F,4U]
  const float* Wh = (const float*)d_in[2];  // [U,4U]
  const float* b  = (const float*)d_in[3];  // [4U]
  const float* Wo = (const float*)d_in[4];  // [U,O]
  const float* bo = (const float*)d_in[5];  // [O]
  // d_in[6] = seq_lengths: does not enter the returned math (see reference docstring)
  float* out = (float*)d_out;               // [B,T,O]

  unsigned short* Wc  = (unsigned short*)d_ws;                               // 768 KB
  unsigned short* Wot = (unsigned short*)((char*)d_ws + (size_t)G_ * K_ * 2); // 16 KB
  unsigned short* xbf = (unsigned short*)((char*)d_ws + (1u << 20));          // 64 MB

  pack_wc_kernel<<<(G_ * K_ + 255) / 256, 256, 0, stream>>>(Wx, Wh, Wc);
  pack_wo_kernel<<<(O_ * U_ + 255) / 256, 256, 0, stream>>>(Wo, Wot);
  pack_x_kernel<<<(int)(((size_t)B_ * T_ * F_ / 4 + 255) / 256), 256, 0, stream>>>(x, xbf);
  lstm_scan_kernel<<<B_ / 16, 256, 0, stream>>>(xbf, Wc, Wot, b, bo, out);
}
